// RNNPolicy_61005715473164
// MI455X (gfx1250) — matrix operand order repaired and checked by reading, not compiled, not run on hardware
//
#include <hip/hip_runtime.h>
#include <hip/hip_bf16.h>
#include <stdint.h>

// ---------------------------------------------------------------------------
// Types for CDNA5 WMMA (wave32, 16x16x32 bf16 -> f32)
// ---------------------------------------------------------------------------
typedef __bf16 bf16_t;
typedef __attribute__((ext_vector_type(16))) __bf16 v16bf;
typedef __attribute__((ext_vector_type(8)))  __bf16 v8bf;
typedef __attribute__((ext_vector_type(4)))  __bf16 v4bf;
typedef __attribute__((ext_vector_type(8)))  float  v8f;
typedef __attribute__((ext_vector_type(4)))  float  v4f;

#define BLK_M 128
#define BLK_N 64
#define BLK_K 32
#define LDSS  40   // LDS row stride in bf16 elems: 80 bytes (16B aligned rows)

__device__ __forceinline__ v4f ld4(const float* p) {
    v4f v; __builtin_memcpy(&v, p, 16); return v;   // tolerate unaligned rows
}

// ---------------------------------------------------------------------------
// Generic GEMM:  C[M=256, N] = act(A[M,K] * W[N,K]^T + bias)
// fp32 in memory, bf16 WMMA math, LDS double-buffered.
// grid = (ceil(N/64), 2), block = 256 threads (8 waves).
// ---------------------------------------------------------------------------
__global__ __launch_bounds__(256)
void k_gemm_bf16(const float* __restrict__ A, int lda,
                 const float* __restrict__ W, int ldw,
                 const float* __restrict__ bias,
                 float* __restrict__ C, int ldc,
                 int N, int K, int do_relu)
{
    __shared__ __align__(16) bf16_t lsA[2][BLK_M * LDSS];
    __shared__ __align__(16) bf16_t lsW[2][BLK_N * LDSS];

    const int tid = threadIdx.x;
    const int m0  = blockIdx.y * BLK_M;
    const int n0  = blockIdx.x * BLK_N;
    const int nk  = (K + BLK_K - 1) / BLK_K;

    v4f avr[4];   // staged A: 128x32 f32 / 256 thr = 4 float4 each
    v4f wvr[2];   // staged W: 64x32 f32 / 256 thr = 2 float4 each

    auto load_regs = [&](int kt) {
        const int k0 = kt * BLK_K;
#pragma unroll
        for (int i = 0; i < 4; ++i) {
            int f   = tid + 256 * i;          // 0..1023
            int row = f >> 3;                 // tile row 0..127
            int c4  = (f & 7) << 2;           // tile col 0..28 step 4
            const float* p = A + (size_t)(m0 + row) * lda + (k0 + c4);
            v4f v = {0.f, 0.f, 0.f, 0.f};
            if (k0 + c4 + 4 <= K) v = ld4(p);
            else {
#pragma unroll
                for (int e = 0; e < 4; ++e) if (k0 + c4 + e < K) v[e] = p[e];
            }
            avr[i] = v;
        }
#pragma unroll
        for (int i = 0; i < 2; ++i) {
            int f   = tid + 256 * i;          // 0..511
            int row = f >> 3;                 // tile row 0..63 (N dim)
            int c4  = (f & 7) << 2;
            int n   = n0 + row;
            v4f v = {0.f, 0.f, 0.f, 0.f};
            if (n < N) {
                const float* p = W + (size_t)n * ldw + (k0 + c4);
                if (k0 + c4 + 4 <= K) v = ld4(p);
                else {
#pragma unroll
                    for (int e = 0; e < 4; ++e) if (k0 + c4 + e < K) v[e] = p[e];
                }
                // speculative prefetch of the weight stream two K-tiles ahead
                __builtin_prefetch(p + 2 * BLK_K, 0, 1);
            }
            wvr[i] = v;
        }
    };

    auto store_lds = [&](int buf) {
#pragma unroll
        for (int i = 0; i < 4; ++i) {
            int f = tid + 256 * i, row = f >> 3, c4 = (f & 7) << 2;
            v4bf b = { (bf16_t)avr[i][0], (bf16_t)avr[i][1],
                       (bf16_t)avr[i][2], (bf16_t)avr[i][3] };
            *(v4bf*)&lsA[buf][row * LDSS + c4] = b;
        }
#pragma unroll
        for (int i = 0; i < 2; ++i) {
            int f = tid + 256 * i, row = f >> 3, c4 = (f & 7) << 2;
            v4bf b = { (bf16_t)wvr[i][0], (bf16_t)wvr[i][1],
                       (bf16_t)wvr[i][2], (bf16_t)wvr[i][3] };
            *(v4bf*)&lsW[buf][row * LDSS + c4] = b;
        }
    };

    const int lane = tid & 31;
    const int wid  = tid >> 5;
    const int wm   = (wid & 3) * 32;    // 4 waves along M (32 rows each)
    const int wn   = (wid >> 2) * 32;   // 2 waves along N (32 cols each)
    const int half = lane >> 4;         // 0: lanes 0-15, 1: lanes 16-31
    const int lrow = lane & 15;

    v8f acc[2][2] = {};

    load_regs(0);
    store_lds(0);
    __syncthreads();

    for (int kt = 0; kt < nk; ++kt) {
        const int cur = kt & 1;
        if (kt + 1 < nk) load_regs(kt + 1);

        const bf16_t* As = lsA[cur];
        const bf16_t* Ws = lsW[cur];

        v16bf afr[2], bfr[2];
        // A fragment layout (ISA 7.12.2, 16-bit A 16x32): lanes 0-15 hold
        // K {0..7,16..23}, lanes 16-31 hold K {8..15,24..31}, row = lane&15.
#pragma unroll
        for (int i = 0; i < 2; ++i) {
            const bf16_t* base = As + (wm + i * 16 + lrow) * LDSS;
            v8bf lo = *(const v8bf*)(base + half * 8);
            v8bf hi = *(const v8bf*)(base + 16 + half * 8);
            afr[i] = __builtin_shufflevector(lo, hi,
                     0,1,2,3,4,5,6,7,8,9,10,11,12,13,14,15);
        }
        // B fragment (32x16): lane holds column n = lane&15, lanes 0-15 hold
        // K {0..7,16..23}, lanes 16-31 hold K {8..15,24..31}.
#pragma unroll
        for (int j = 0; j < 2; ++j) {
            const bf16_t* base = Ws + (wn + j * 16 + lrow) * LDSS;
            v8bf lo = *(const v8bf*)(base + half * 8);
            v8bf hi = *(const v8bf*)(base + 16 + half * 8);
            bfr[j] = __builtin_shufflevector(lo, hi,
                     0,1,2,3,4,5,6,7,8,9,10,11,12,13,14,15);
        }

#pragma unroll
        for (int i = 0; i < 2; ++i)
#pragma unroll
            for (int j = 0; j < 2; ++j)
                acc[i][j] = __builtin_amdgcn_wmma_f32_16x16x32_bf16(
                    /*neg_a=*/false, afr[i], /*neg_b=*/false, bfr[j],
                    /*c_mod=*/(short)0, acc[i][j],
                    /*reuse_a=*/false, /*reuse_b=*/false);

        if (kt + 1 < nk) store_lds(cur ^ 1);
        __syncthreads();
    }

    // Epilogue: C/D layout — VGPR r of lane l holds (M = r + 8*(l>>4), N = l&15).
#pragma unroll
    for (int i = 0; i < 2; ++i) {
#pragma unroll
        for (int j = 0; j < 2; ++j) {
            int col = n0 + wn + j * 16 + lrow;
            if (col < N) {
                float bs = bias ? bias[col] : 0.f;
                int row0 = m0 + wm + i * 16 + half * 8;
#pragma unroll
                for (int r = 0; r < 8; ++r) {
                    float v = acc[i][j][r] + bs;
                    if (do_relu) v = fmaxf(v, 0.f);
                    C[(size_t)(row0 + r) * ldc + col] = v;
                }
            }
        }
    }
}

// ---------------------------------------------------------------------------
// Small supporting kernels
// ---------------------------------------------------------------------------
__global__ void k_init(const float* __restrict__ b_ih, const float* __restrict__ b_hh,
                       float* __restrict__ bsum, float* __restrict__ entropy,
                       float* __restrict__ logp)
{
    int i = blockIdx.x * blockDim.x + threadIdx.x;
    if (i < 2048) bsum[i] = b_ih[i] + b_hh[i];
    if (i < 256)  logp[i] = 0.f;
    if (i == 0)   *entropy = 0.f;
}

__global__ void k_concat(const float* __restrict__ a, const float* __restrict__ o,
                         float* __restrict__ x)
{
    int i = blockIdx.x * blockDim.x + threadIdx.x;  // 256*6144
    int b = i / 6144, c = i % 6144;
    x[i] = (c < 3072) ? a[b * 3072 + c] : o[b * 3072 + (c - 3072)];
}

__global__ void k_embed(const int* __restrict__ last_pos, const int* __restrict__ ep,
                        const float* __restrict__ emb_end, const float* __restrict__ emb_ep,
                        float* __restrict__ feats)
{
    int b = blockIdx.x, t = threadIdx.x;
    if (t < 16)       feats[(size_t)b * 4632 + 4608 + t] = emb_end[last_pos[b] * 16 + t];
    else if (t < 24)  feats[(size_t)b * 4632 + 4624 + (t - 16)] = emb_ep[ep[b] * 8 + (t - 16)];
}

__global__ void k_lstm(const float* __restrict__ gates, float* __restrict__ z /*ldz=528*/)
{
    int idx = blockIdx.x * blockDim.x + threadIdx.x;   // 256*512
    if (idx >= 256 * 512) return;
    int b = idx >> 9, j = idx & 511;
    float gi = gates[(size_t)b * 2048 + j];
    float gg = gates[(size_t)b * 2048 + 1024 + j];
    float go = gates[(size_t)b * 2048 + 1536 + j];
    float si = 1.f / (1.f + __expf(-gi));
    float so = 1.f / (1.f + __expf(-go));
    float c  = si * tanhf(gg);
    z[(size_t)b * 528 + j] = so * tanhf(c);
}

__device__ __forceinline__ uint32_t wang_hash(uint32_t s)
{
    s = (s ^ 61u) ^ (s >> 16); s *= 9u; s ^= s >> 4;
    s *= 0x27d4eb2du; s ^= s >> 15; return s;
}

// One block per batch row; log-softmax + entropy + Gumbel-max sample.
__global__ void k_sample(const float* __restrict__ logits, int k,
                         int* __restrict__ actions, float* __restrict__ logp_acc,
                         float* __restrict__ entropy, const float* __restrict__ emb_dec,
                         float* __restrict__ z /*ldz=528, writes cols 512..527*/)
{
    int b = blockIdx.x, t = threadIdx.x;   // 64 threads
    __shared__ float sl[64];
    __shared__ int   s_act;
    sl[t] = logits[(size_t)b * 64 + t];
    __syncthreads();
    if (t == 0) {
        float m = -1e30f;
        for (int i = 0; i < 64; ++i) m = fmaxf(m, sl[i]);
        float s = 0.f;
        for (int i = 0; i < 64; ++i) s += __expf(sl[i] - m);
        float logZ = m + __logf(s);
        float ent = 0.f, best = -1e30f; int bi = 0;
        for (int i = 0; i < 64; ++i) {
            float lp = sl[i] - logZ;
            ent += -__expf(lp) * lp;
            uint32_t h = wang_hash(0x9e3779b9u ^ (uint32_t)(((b * 4 + k) << 8) | i));
            float u = (float)(h >> 8) * (1.0f / 16777216.0f) + 1e-12f;
            float g = -__logf(-__logf(u));
            float v = sl[i] + g;
            if (v > best) { best = v; bi = i; }
        }
        s_act = bi;
        actions[b * 4 + k] = bi;
        logp_acc[b] += sl[bi] - logZ;
        atomicAdd(entropy, ent);
    }
    __syncthreads();
    if (t < 16)
        z[(size_t)b * 528 + 512 + t] = emb_dec[(size_t)k * 64 * 16 + s_act * 16 + t];
}

// Pack outputs: actions[256*4] (as float), entropy[1], log_probabilities[256]
__global__ void k_final(const int* __restrict__ actions, const float* __restrict__ entropy,
                        const float* __restrict__ logp, float* __restrict__ out)
{
    int i = blockIdx.x * blockDim.x + threadIdx.x;
    if (i < 1024)       out[i] = (float)actions[i];
    else if (i == 1024) out[1024] = *entropy;
    else if (i < 1281)  out[i] = logp[i - 1025];
}

// ---------------------------------------------------------------------------
// Launcher
// ---------------------------------------------------------------------------
static inline void launch_gemm(hipStream_t stream,
                               const float* A, int lda, const float* W, int ldw,
                               const float* bias, float* C, int ldc,
                               int N, int K, int relu)
{
    dim3 grid((N + BLK_N - 1) / BLK_N, 256 / BLK_M);
    k_gemm_bf16<<<grid, dim3(256), 0, stream>>>(A, lda, W, ldw, bias, C, ldc, N, K, relu);
}

extern "C" void kernel_launch(void* const* d_in, const int* in_sizes, int n_in,
                              void* d_out, int out_size, void* d_ws, size_t ws_size,
                              hipStream_t stream)
{
    const float* actual    = (const float*)d_in[0];
    const float* objective = (const float*)d_in[1];
    const int*   last_pos  = (const int*)d_in[2];
    const int*   ep_pct    = (const int*)d_in[3];
    const float* Wfe  = (const float*)d_in[4];
    const float* bfe  = (const float*)d_in[5];
    const float* emb_end = (const float*)d_in[6];
    const float* emb_ep  = (const float*)d_in[7];
    const float* W1 = (const float*)d_in[8];  const float* b1 = (const float*)d_in[9];
    const float* W2 = (const float*)d_in[10]; const float* b2 = (const float*)d_in[11];
    const float* W3 = (const float*)d_in[12]; const float* b3 = (const float*)d_in[13];
    const float* W_ih = (const float*)d_in[14];
    const float* b_ih = (const float*)d_in[16];
    const float* b_hh = (const float*)d_in[17];
    const float* Wd1 = (const float*)d_in[18]; const float* bd1 = (const float*)d_in[19];
    const float* Wd2 = (const float*)d_in[20]; const float* bd2 = (const float*)d_in[21];
    const float* emb_dec = (const float*)d_in[22];
    const float* Wout = (const float*)d_in[23]; const float* bout = (const float*)d_in[24];

    float* ws     = (float*)d_ws;
    float* x      = ws;                                // 256*6144
    float* feats  = x      + (size_t)256 * 6144;       // 256*4632
    float* h1     = feats  + (size_t)256 * 4632;       // 256*2223
    float* h2     = h1     + (size_t)256 * 2223;       // 256*1067
    float* h3     = h2     + (size_t)256 * 1067;       // 256*512
    float* gates  = h3     + (size_t)256 * 512;        // 256*2048
    float* bsum   = gates  + (size_t)256 * 2048;       // 2048
    float* z0     = bsum   + 2048;                     // 256*528
    float* z1     = z0     + (size_t)256 * 528;        // 256*528
    float* hm     = z1     + (size_t)256 * 528;        // 256*256
    float* logits = hm     + (size_t)256 * 256;        // 256*64
    float* logp   = logits + (size_t)256 * 64;         // 256
    float* entrop = logp   + 256;                      // 1 (+pad)
    int*   acts   = (int*)(entrop + 4);                // 1024 ints

    k_init  <<<8,    256, 0, stream>>>(b_ih, b_hh, bsum, entrop, logp);
    k_concat<<<6144, 256, 0, stream>>>(actual, objective, x);

    // fv = relu(x @ Wfe^T + bfe) written directly into feats[:, :4608]
    launch_gemm(stream, x, 6144, Wfe, 6144, bfe, feats, 4632, 4608, 6144, 1);
    k_embed<<<256, 32, 0, stream>>>(last_pos, ep_pct, emb_end, emb_ep, feats);

    launch_gemm(stream, feats, 4632, W1, 4632, b1, h1, 2223, 2223, 4632, 1);
    launch_gemm(stream, h1,    2223, W2, 2223, b2, h2, 1067, 1067, 2223, 1);
    launch_gemm(stream, h2,    1067, W3, 1067, b3, h3, 512,  512,  1067, 1);
    launch_gemm(stream, h3,    512,  W_ih, 512, bsum, gates, 2048, 2048, 512, 0);

    k_lstm<<<512, 256, 0, stream>>>(gates, z0);

    float* zc = z0; float* zn = z1;
    for (int k = 0; k < 4; ++k) {
        launch_gemm(stream, zc, 528, Wd1 + (size_t)k * 256 * 512, 512,
                    bd1 + k * 256, hm, 256, 256, 512, 1);
        launch_gemm(stream, hm, 256, Wd2 + (size_t)k * 64 * 256, 256,
                    bd2 + k * 64, logits, 64, 64, 256, 0);
        k_sample<<<256, 64, 0, stream>>>(logits, k, acts, logp, entrop, emb_dec, zc);
        launch_gemm(stream, zc, 528, Wout, 528, bout, zn, 528, 512, 528, 1);
        float* t = zc; zc = zn; zn = t;
    }

    k_final<<<6, 256, 0, stream>>>(acts, entrop, logp, (float*)d_out);
}
